// TypedLinear_30562987278726
// MI455X (gfx1250) — matrix operand order repaired and checked by reading, not compiled, not run on hardware
//
#include <hip/hip_runtime.h>

// Problem constants (match reference)
#define NROWS  65536
#define KIN    256
#define MOUT   256
#define NTYPES 8
#define PERM_CAP  (NROWS + NTYPES * 16)        // 16-aligned per-type segments
#define ROW_TILES (NROWS / 16 + NTYPES)        // fixed upper bound on 16-row tiles
#define WELEMS    (NTYPES * MOUT * KIN)        // 524288 weights

// Workspace layout (bytes):
//   [0)        int counts[8]
//   [32)       int cursors[8]
//   [64)       int off[9]
//   [128)      int perm[PERM_CAP]
//   [263168)   __bf16 Whi[WELEMS]   (1 MB)
//   [+1048576) __bf16 Wlo[WELEMS]   (1 MB)
#define WS_COUNTS  0
#define WS_CURSORS 32
#define WS_OFF     64
#define WS_PERM    128
#define WS_WHI     263168
#define WS_WLO     (263168 + WELEMS * 2)

typedef __attribute__((ext_vector_type(16))) __bf16 v16bf;
typedef __attribute__((ext_vector_type(8)))  __bf16 v8bf;
typedef __attribute__((ext_vector_type(8)))  float  v8f;

// ---------------- binning kernels ----------------

__global__ void k_init(int* cnt_cur) {
    if (threadIdx.x < 2 * NTYPES) cnt_cur[threadIdx.x] = 0;
}

__global__ void k_hist(const int* __restrict__ types, int* __restrict__ counts) {
    int i = blockIdx.x * 256 + threadIdx.x;
    if (i < NROWS) atomicAdd(&counts[types[i]], 1);
}

__global__ void k_scan(const int* __restrict__ counts, int* __restrict__ off) {
    if (threadIdx.x == 0) {
        int a = 0;
        off[0] = 0;
        for (int t = 0; t < NTYPES; ++t) {
            a += (counts[t] + 15) & ~15;   // 16-align each type segment
            off[t + 1] = a;
        }
    }
}

__global__ void k_fill(int* __restrict__ perm) {
    int i = blockIdx.x * 256 + threadIdx.x;
    if (i < PERM_CAP) perm[i] = -1;
}

__global__ void k_scatter(const int* __restrict__ types, const int* __restrict__ off,
                          int* __restrict__ cursors, int* __restrict__ perm) {
    int i = blockIdx.x * 256 + threadIdx.x;
    if (i < NROWS) {
        int t = types[i];
        int p = off[t] + atomicAdd(&cursors[t], 1);
        perm[p] = i;
    }
}

// Split W into bf16 hi/lo once per launch (L2-resident, reused by all tiles)
__global__ void k_convw(const float* __restrict__ W,
                        __bf16* __restrict__ Whi, __bf16* __restrict__ Wlo) {
    int i = blockIdx.x * 256 + threadIdx.x;
    if (i < WELEMS) {
        float f = W[i];
        __bf16 h = (__bf16)f;
        Whi[i] = h;
        Wlo[i] = (__bf16)(f - (float)h);
    }
}

// ---------------- grouped GEMM via bf16x3 WMMA ----------------

__global__ __launch_bounds__(256)
void k_gemm(const float* __restrict__ x, const float* __restrict__ bias,
            float* __restrict__ out, const int* __restrict__ off,
            const int* __restrict__ perm,
            const __bf16* __restrict__ Whi, const __bf16* __restrict__ Wlo) {
    // X tile staged as bf16 hi/lo; row padded +8 bf16 so the 16B row chunks
    // rotate across LDS banks (stride = 528 B -> +4 banks/row, 16 rows cover 64 banks)
    __shared__ __align__(16) __bf16 sHi[16][KIN + 8];
    __shared__ __align__(16) __bf16 sLo[16][KIN + 8];
    __shared__ int sRows[16];

    const int tile = blockIdx.x;
    const int base = tile * 16;
    if (base >= off[NTYPES]) return;          // uniform exit, before any barrier

    // locate this tile's type segment (segments are 16-aligned)
    int t = 0;
    #pragma unroll
    for (int i = 1; i < NTYPES; ++i)
        if (base >= off[i]) t = i;

    const int tid = threadIdx.x;

    // ---- stage 16 gathered rows of x into LDS as bf16 hi/lo ----
    {
        const int row  = tid >> 4;            // 0..15
        const int cseg = (tid & 15) * 16;     // 16 floats per thread
        const int ridx = perm[base + row];
        if (tid < 16) sRows[tid] = perm[base + tid];
        if (ridx >= 0) {
            const float* xr = x + (size_t)ridx * KIN + cseg;
            #pragma unroll
            for (int j = 0; j < 16; ++j) {
                float  f = xr[j];
                __bf16 h = (__bf16)f;
                sHi[row][cseg + j] = h;
                sLo[row][cseg + j] = (__bf16)(f - (float)h);
            }
        } else {
            #pragma unroll
            for (int j = 0; j < 16; ++j) {
                sHi[row][cseg + j] = (__bf16)0.0f;
                sLo[row][cseg + j] = (__bf16)0.0f;
            }
        }
    }
    __syncthreads();

    // ---- 8 waves x 2 column tiles = 256 output columns ----
    const int lane = tid & 31;
    const int wid  = tid >> 5;                // 0..7
    const int m    = lane & 15;               // A row / D col index
    const int h    = lane >> 4;               // lane half

    #pragma unroll
    for (int sub = 0; sub < 2; ++sub) {
        const int n  = m;                     // D column within tile
        const int c0 = (wid * 2 + sub) * 16;  // output column base
        const __bf16* wh = Whi + ((size_t)t * MOUT + c0 + n) * KIN;
        const __bf16* wl = Wlo + ((size_t)t * MOUT + c0 + n) * KIN;

        v8f acc = {};
        #pragma unroll
        for (int k0 = 0; k0 < KIN; k0 += 32) {
            // A layout (16-bit A 16x32): elems 0..15 = K[k0+16h .. k0+16h+15]
            v8bf a0h = *(const v8bf*)&sHi[m][k0 + 16 * h];
            v8bf a1h = *(const v8bf*)&sHi[m][k0 + 16 * h + 8];
            v8bf a0l = *(const v8bf*)&sLo[m][k0 + 16 * h];
            v8bf a1l = *(const v8bf*)&sLo[m][k0 + 16 * h + 8];
            v16bf ahi = __builtin_shufflevector(a0h, a1h,
                         0,1,2,3,4,5,6,7,8,9,10,11,12,13,14,15);
            v16bf alo = __builtin_shufflevector(a0l, a1l,
                         0,1,2,3,4,5,6,7,8,9,10,11,12,13,14,15);
            // B layout (32x16): lane half h holds K = k0+16h .. k0+16h+15 of column c0+n
            v16bf bhi = *(const v16bf*)&wh[k0 + 16 * h];
            v16bf blo = *(const v16bf*)&wl[k0 + 16 * h];

            // bf16x3 split-precision accumulate: a*b ~= ahi*bhi + ahi*blo + alo*bhi
            acc = __builtin_amdgcn_wmma_f32_16x16x32_bf16(false, ahi, false, bhi,
                                                          (short)0, acc, false, false);
            acc = __builtin_amdgcn_wmma_f32_16x16x32_bf16(false, ahi, false, blo,
                                                          (short)0, acc, false, false);
            acc = __builtin_amdgcn_wmma_f32_16x16x32_bf16(false, alo, false, bhi,
                                                          (short)0, acc, false, false);
        }

        const float bv = bias[t * MOUT + c0 + n];
        // D layout: lane covers col n, rows v + 8h, v = 0..7
        #pragma unroll
        for (int v = 0; v < 8; ++v) {
            const int r = sRows[v + 8 * h];
            if (r >= 0) out[(size_t)r * MOUT + c0 + n] = acc[v] + bv;
        }
    }
}

// ---------------- host launch ----------------

extern "C" void kernel_launch(void* const* d_in, const int* in_sizes, int n_in,
                              void* d_out, int out_size, void* d_ws, size_t ws_size,
                              hipStream_t stream) {
    const float* x     = (const float*)d_in[0];
    const int*   types = (const int*)d_in[1];
    const float* W     = (const float*)d_in[2];
    const float* bias  = (const float*)d_in[3];
    float*       out   = (float*)d_out;

    char* ws = (char*)d_ws;
    int*    counts  = (int*)(ws + WS_COUNTS);     // counts+cursors contiguous
    int*    cursors = (int*)(ws + WS_CURSORS);
    int*    off     = (int*)(ws + WS_OFF);
    int*    perm    = (int*)(ws + WS_PERM);
    __bf16* Whi     = (__bf16*)(ws + WS_WHI);
    __bf16* Wlo     = (__bf16*)(ws + WS_WLO);

    k_init   <<<1, 64, 0, stream>>>(counts);                       // counts + cursors
    k_hist   <<<NROWS / 256, 256, 0, stream>>>(types, counts);
    k_scan   <<<1, 32, 0, stream>>>(counts, off);
    k_fill   <<<(PERM_CAP + 255) / 256, 256, 0, stream>>>(perm);
    k_scatter<<<NROWS / 256, 256, 0, stream>>>(types, off, cursors, perm);
    k_convw  <<<WELEMS / 256, 256, 0, stream>>>(W, Whi, Wlo);
    k_gemm   <<<ROW_TILES, 256, 0, stream>>>(x, bias, out, off, perm, Whi, Wlo);
}
